// Model_67851893342546
// MI455X (gfx1250) — compile-verified, matrix operand order checked by reading, not run on hardware
//
#include <hip/hip_runtime.h>
#include <hip/hip_bf16.h>
#include <math.h>

// ---------------------------------------------------------------------------
// Types / helpers
// ---------------------------------------------------------------------------
typedef __bf16 bf16_t;
typedef __attribute__((ext_vector_type(16))) __bf16 v16bf;
typedef __attribute__((ext_vector_type(8)))  __bf16 v8bf;
typedef __attribute__((ext_vector_type(2)))  __bf16 v2bf;
typedef __attribute__((ext_vector_type(8)))  float  v8f;
typedef int gvec4i __attribute__((vector_size(16)));   // matches builtin param

// Async global->LDS (CDNA5) when the toolchain exposes the builtin;
// otherwise plain vectorized copy through VGPRs.
#if defined(__has_builtin)
# if __has_builtin(__builtin_amdgcn_global_load_async_to_lds_b128)
#  define HAVE_ASYNC_LDS 1
# endif
#endif

#ifdef HAVE_ASYNC_LDS
# define ASYNC_COPY_B128(gsrc, ldst)                                        \
    __builtin_amdgcn_global_load_async_to_lds_b128(                         \
        (__attribute__((address_space(1))) gvec4i*)(gsrc),                  \
        (__attribute__((address_space(3))) gvec4i*)(ldst), 0, 0)
# if defined(__has_builtin) && __has_builtin(__builtin_amdgcn_s_wait_asynccnt)
#  define ASYNC_WAIT() __builtin_amdgcn_s_wait_asynccnt(0)
# else
#  define ASYNC_WAIT() asm volatile("s_wait_asynccnt 0" ::: "memory")
# endif
#else
# define ASYNC_COPY_B128(gsrc, ldst) (*(v8bf*)(ldst) = *(const v8bf*)(gsrc))
# define ASYNC_WAIT() ((void)0)
#endif

__device__ __forceinline__ bf16_t to_bf16(float f) {
    unsigned int u = __builtin_bit_cast(unsigned int, f);
    unsigned int r = (u + 0x7FFFu + ((u >> 16) & 1u)) >> 16;
    return __builtin_bit_cast(bf16_t, (unsigned short)r);
}
__device__ __forceinline__ float gelu_f(float x) {
    return 0.5f * x * (1.0f + erff(x * 0.70710678118654752f));
}
// WMMA fragment K-index map (ISA 7.12.2, 16-bit A 16x32):
__device__ __forceinline__ int kmap(int v, int half) {
    return 2 * (v & 3) + 16 * (v >> 2) + 8 * half;
}

// ---------------------------------------------------------------------------
// Generic bf16 WMMA GEMM: C = act(A(MxK) @ B(KxN) + bias), fp32 out.
// 64x64 tile/block; 4 waves (128 thr); wave = 16 rows x 64 cols
// (1 A-frag, 4 B-frags, 4 wmma accumulators per K step).
// Fast path (full tile): 16B vector / async-LDS loads. Edge tiles: clamped
// 32-bit offsets + bit-select (no divergent branches, no 64-bit mul chains).
// ---------------------------------------------------------------------------
__global__ __launch_bounds__(128) void k_gemm(
    const bf16_t* __restrict__ A, const bf16_t* __restrict__ B,
    const float* __restrict__ bias, float* __restrict__ C,
    int M, int N, int K, int lda, int ldb, int ldc, int act)
{
    __shared__ bf16_t As[64][40];   // [row][k]
    __shared__ bf16_t Bts[64][40];  // [n][k] (B transposed)

    const int t = threadIdx.x;
    const int m0 = blockIdx.y * 64, n0 = blockIdx.x * 64;
    const int wave = t >> 5, lane = t & 31;
    const int half = lane >> 4, lr = lane & 15;
    const bool mn_full = (m0 + 64 <= M) && (n0 + 64 <= N);

    v8f acc[4] = {{}, {}, {}, {}};

    for (int k0 = 0; k0 < K; k0 += 32) {
        const bool full = mn_full && (k0 + 32 <= K);
        if (full) {
            // A tile: two 16B chunks per thread (async global->LDS when avail)
            const int ac = (t & 3) * 8;
            #pragma unroll
            for (int rr = 0; rr < 2; rr++) {
                const int ar = (t >> 2) + 32 * rr;
                const bf16_t* gA = &A[(size_t)(m0 + ar) * lda + k0 + ac];
                ASYNC_COPY_B128(gA, &As[ar][ac]);
                if (k0 + 32 < K) __builtin_prefetch((const void*)(gA + 32), 0, 0);
            }
            // B tile: two 16B vector loads, transpose-scatter to LDS
            const int bn = (t & 7) * 8;
            #pragma unroll
            for (int ch = 0; ch < 2; ch++) {
                const int bk = (t >> 3) + 16 * ch;
                v8bf bv = *(const v8bf*)&B[(size_t)(k0 + bk) * ldb + n0 + bn];
                #pragma unroll
                for (int i = 0; i < 8; i++) Bts[bn + i][bk] = bv[i];
                if (k0 + 32 < K)
                    __builtin_prefetch((const void*)&B[(size_t)(k0 + 32 + bk) * ldb + n0 + bn], 0, 0);
            }
            ASYNC_WAIT();
        } else {
            // clamped 32-bit offsets + bit-select (branch-free OOB handling)
            #pragma unroll
            for (int i = 0; i < 16; i++) {
                int e = t * 16 + i; int r = e >> 5, c = e & 31;
                int gr = m0 + r, gc = k0 + c;
                bool ok = (gr < M) && (gc < K);
                int addr = ok ? (gr * lda + gc) : 0;
                unsigned short u = __builtin_bit_cast(unsigned short, A[addr]);
                As[r][c] = __builtin_bit_cast(bf16_t, ok ? u : (unsigned short)0);
            }
            #pragma unroll
            for (int i = 0; i < 16; i++) {
                int e = t * 16 + i; int kk = e >> 6, n = e & 63;
                int gk = k0 + kk, gn = n0 + n;
                bool ok = (gk < K) && (gn < N);
                int addr = ok ? (gk * ldb + gn) : 0;
                unsigned short u = __builtin_bit_cast(unsigned short, B[addr]);
                Bts[n][kk] = __builtin_bit_cast(bf16_t, ok ? u : (unsigned short)0);
            }
        }
        __syncthreads();

        v16bf a;
        {
            int row = 16 * wave + lr;
            #pragma unroll
            for (int v = 0; v < 8; v++) {
                int k = kmap(v, half);
                v2bf p = *(const v2bf*)&As[row][k];
                a[2 * v] = p.x; a[2 * v + 1] = p.y;
            }
        }
        #pragma unroll
        for (int j = 0; j < 4; j++) {
            v16bf b;
            int n = 16 * j + lr;
            #pragma unroll
            for (int v = 0; v < 8; v++) {
                int k = kmap(v, half);
                v2bf p = *(const v2bf*)&Bts[n][k];
                b[2 * v] = p.x; b[2 * v + 1] = p.y;
            }
            acc[j] = __builtin_amdgcn_wmma_f32_16x16x32_bf16(false, a, false, b,
                        (short)0, acc[j], false, false);
        }
        __syncthreads();
    }

    #pragma unroll
    for (int j = 0; j < 4; j++) {
        int col = n0 + 16 * j + lr;
        if (col < N) {
            float bv = bias ? bias[col] : 0.0f;
            #pragma unroll
            for (int v = 0; v < 8; v++) {
                int row = m0 + 16 * wave + v + 8 * half;
                if (row < M) {
                    float x = acc[j][v] + bv;
                    if (act == 1) x = gelu_f(x);
                    C[(size_t)row * ldc + col] = x;
                }
            }
        }
    }
}

// ---------------------------------------------------------------------------
// Flash attention: one (64-query block, head) per 128-thread block (4 waves).
// dh = 32 (== one bf16 WMMA K step). Online softmax, O accumulated in regs.
// ---------------------------------------------------------------------------
__global__ __launch_bounds__(128) void k_flash(
    const bf16_t* __restrict__ Qg, const bf16_t* __restrict__ Kg,
    const bf16_t* __restrict__ Vg, float* __restrict__ O, int M, int D)
{
    __shared__ bf16_t Qs[64][40];
    __shared__ bf16_t Ks[64][40];   // [key][k]  (== B^T for Q@K^T)
    __shared__ bf16_t Vt[32][72];   // [d][key]  (== B^T for P@V)
    __shared__ bf16_t Ps[64][72];   // [qrow][key]

    const int qb = blockIdx.x, h = blockIdx.y;
    const int t = threadIdx.x, wave = t >> 5, lane = t & 31;
    const int half = lane >> 4, lr = lane & 15;
    const int hoff = h * 32;
    const float scale = 0.17677669529663687f; // 1/sqrt(32)

    #pragma unroll
    for (int e = t; e < 256; e += 128) {       // 64 rows x 4 chunks of 16B
        int r = e >> 2, c = (e & 3) * 8;
        ASYNC_COPY_B128(&Qg[(size_t)(qb * 64 + r) * D + hoff + c], &Qs[r][c]);
    }
    ASYNC_WAIT();
    __syncthreads();

    v16bf aq;
    {
        int row = 16 * wave + lr;
        #pragma unroll
        for (int v = 0; v < 8; v++) {
            int k = kmap(v, half);
            v2bf p = *(const v2bf*)&Qs[row][k];
            aq[2 * v] = p.x; aq[2 * v + 1] = p.y;
        }
    }

    float m_run[8], l_run[8];
    #pragma unroll
    for (int v = 0; v < 8; v++) { m_run[v] = -1e30f; l_run[v] = 0.0f; }
    v8f o0 = {}, o1 = {};

    const int nkb = M / 64;
    for (int kb = 0; kb < nkb; kb++) {
        __syncthreads();
        #pragma unroll
        for (int e = t; e < 256; e += 128) {
            int r = e >> 2, c = (e & 3) * 8;
            ASYNC_COPY_B128(&Kg[(size_t)(kb * 64 + r) * D + hoff + c], &Ks[r][c]);
            v8bf vv = *(const v8bf*)&Vg[(size_t)(kb * 64 + r) * D + hoff + c];
            #pragma unroll
            for (int i = 0; i < 8; i++) Vt[c + i][r] = vv[i];
        }
        ASYNC_WAIT();
        __syncthreads();

        // S = Q @ K^T  (4 N-tiles of 16 keys, single K-step of 32)
        v8f s[4];
        #pragma unroll
        for (int j = 0; j < 4; j++) {
            v16bf bb; int n = 16 * j + lr;
            #pragma unroll
            for (int v = 0; v < 8; v++) {
                int k = kmap(v, half);
                v2bf p = *(const v2bf*)&Ks[n][k];
                bb[2 * v] = p.x; bb[2 * v + 1] = p.y;
            }
            v8f z = {};
            s[j] = __builtin_amdgcn_wmma_f32_16x16x32_bf16(false, aq, false, bb,
                    (short)0, z, false, false);
        }

        // online softmax per query row (row spread over 16 lanes of a half)
        #pragma unroll
        for (int v = 0; v < 8; v++) {
            float mx = -1e30f;
            #pragma unroll
            for (int j = 0; j < 4; j++) {
                float sv = s[j][v] * scale;
                s[j][v] = sv;
                mx = fmaxf(mx, sv);
            }
            for (int off = 1; off < 16; off <<= 1) mx = fmaxf(mx, __shfl_xor(mx, off));
            float mn = fmaxf(m_run[v], mx);
            float alpha = __expf(m_run[v] - mn);
            float rs = 0.0f;
            #pragma unroll
            for (int j = 0; j < 4; j++) {
                float p = __expf(s[j][v] - mn);
                s[j][v] = p;
                rs += p;
            }
            for (int off = 1; off < 16; off <<= 1) rs += __shfl_xor(rs, off);
            l_run[v] = l_run[v] * alpha + rs;
            m_run[v] = mn;
            o0[v] *= alpha; o1[v] *= alpha;
            int prow = 16 * wave + v + 8 * half;
            #pragma unroll
            for (int j = 0; j < 4; j++) Ps[prow][16 * j + lr] = to_bf16(s[j][v]);
        }

        // O += P @ V  (K = 64 keys -> 2 steps; N = dh 32 -> 2 tiles)
        #pragma unroll
        for (int ks = 0; ks < 2; ks++) {
            v16bf ap; int row = 16 * wave + lr;
            #pragma unroll
            for (int v = 0; v < 8; v++) {
                int k = kmap(v, half) + 32 * ks;
                v2bf p = *(const v2bf*)&Ps[row][k];
                ap[2 * v] = p.x; ap[2 * v + 1] = p.y;
            }
            #pragma unroll
            for (int jo = 0; jo < 2; jo++) {
                v16bf bb; int n = 16 * jo + lr;
                #pragma unroll
                for (int v = 0; v < 8; v++) {
                    int k = kmap(v, half) + 32 * ks;
                    v2bf p = *(const v2bf*)&Vt[n][k];
                    bb[2 * v] = p.x; bb[2 * v + 1] = p.y;
                }
                if (jo == 0)
                    o0 = __builtin_amdgcn_wmma_f32_16x16x32_bf16(false, ap, false, bb,
                            (short)0, o0, false, false);
                else
                    o1 = __builtin_amdgcn_wmma_f32_16x16x32_bf16(false, ap, false, bb,
                            (short)0, o1, false, false);
            }
        }
    }

    #pragma unroll
    for (int v = 0; v < 8; v++) {
        int row = qb * 64 + 16 * wave + v + 8 * half;
        float inv = 1.0f / l_run[v];
        O[(size_t)row * D + hoff + lr]      = o0[v] * inv;
        O[(size_t)row * D + hoff + 16 + lr] = o1[v] * inv;
    }
}

// ---------------------------------------------------------------------------
// RevIN stats + patch A-matrix build (bf16). One block per (b,c) series.
// ---------------------------------------------------------------------------
__global__ __launch_bounds__(256) void k_prep(
    const float* __restrict__ x, float* __restrict__ meanv,
    float* __restrict__ stdv, bf16_t* __restrict__ patchA)
{
    const int bc = blockIdx.x;      // 0..671
    const int b = bc / 21, c = bc % 21;
    const int t = threadIdx.x;
    __shared__ float red[256];

    float s = 0.0f;
    for (int i = t; i < 512; i += 256) s += x[((size_t)b * 512 + i) * 21 + c];
    red[t] = s; __syncthreads();
    for (int o = 128; o > 0; o >>= 1) { if (t < o) red[t] += red[t + o]; __syncthreads(); }
    float mu = red[0] / 512.0f;
    __syncthreads();

    float vv = 0.0f;
    for (int i = t; i < 512; i += 256) {
        float d = x[((size_t)b * 512 + i) * 21 + c] - mu;
        vv += d * d;
    }
    red[t] = vv; __syncthreads();
    for (int o = 128; o > 0; o >>= 1) { if (t < o) red[t] += red[t + o]; __syncthreads(); }
    float sd = sqrtf(red[0] / 512.0f + 1e-5f);
    if (t == 0) { meanv[bc] = mu; stdv[bc] = sd; }

    for (int e = t; e < 64 * 16; e += 256) {
        int p = e >> 4, k = e & 15;
        int tt = p * 8 + k; if (tt > 511) tt = 511;     // replicate-pad
        float xv = (x[((size_t)b * 512 + tt) * 21 + c] - mu) / sd;
        patchA[((size_t)bc * 64 + p) * 16 + k] = to_bf16(xv);
    }
}

// add sinusoidal PE (N=64 patches, D=256) in place
__global__ void k_add_pe(float* __restrict__ p, int total)
{
    int i = blockIdx.x * 256 + threadIdx.x;
    if (i >= total) return;
    int col = i & 255, row = i >> 8, n = row & 63;
    float div = __expf(-0.035977891f * (float)(col & ~1)); // ln(1e4)/256
    float ang = (float)n * div;
    p[i] += (col & 1) ? cosf(ang) : sinf(ang);
}

__global__ void k_f32_to_bf16(const float* __restrict__ s, bf16_t* __restrict__ d, int n)
{
    int i = blockIdx.x * 256 + threadIdx.x;
    if (i < n) d[i] = to_bf16(s[i]);
}

// scorer: LN(dim=64) then dot with W2 + b2. One 64-thread block per row.
__global__ __launch_bounds__(64) void k_scorer(
    const float* __restrict__ h, const float* __restrict__ g,
    const float* __restrict__ bta, const float* __restrict__ w2,
    const float* __restrict__ b2, float* __restrict__ scores)
{
    const int row = blockIdx.x, t = threadIdx.x;
    __shared__ float red[64];
    float v = h[(size_t)row * 64 + t];
    red[t] = v; __syncthreads();
    for (int o = 32; o > 0; o >>= 1) { if (t < o) red[t] += red[t + o]; __syncthreads(); }
    float mu = red[0] / 64.0f; __syncthreads();
    red[t] = (v - mu) * (v - mu); __syncthreads();
    for (int o = 32; o > 0; o >>= 1) { if (t < o) red[t] += red[t + o]; __syncthreads(); }
    float inv = rsqrtf(red[0] / 64.0f + 1e-5f); __syncthreads();
    float ln = (v - mu) * inv * g[t] + bta[t];
    red[t] = ln * w2[t]; __syncthreads();
    for (int o = 32; o > 0; o >>= 1) { if (t < o) red[t] += red[t + o]; __syncthreads(); }
    if (t == 0) scores[row] = red[0] + b2[0];
}

// top-6 of 64 (first-occurrence ties like lax.top_k), sorted ascending
__global__ void k_topk(const float* __restrict__ scores, int* __restrict__ idx)
{
    int bc = blockIdx.x * blockDim.x + threadIdx.x;
    if (bc >= 672) return;
    const float* s = scores + (size_t)bc * 64;
    unsigned long long mask = 0ull;
    int sel[6];
    for (int j = 0; j < 6; j++) {
        float best = -1e30f; int bi = 0;
        for (int i = 0; i < 64; i++) {
            if (!((mask >> i) & 1ull) && s[i] > best) { best = s[i]; bi = i; }
        }
        mask |= 1ull << bi;
        sel[j] = bi;
    }
    for (int a = 1; a < 6; a++) {          // insertion sort ascending
        int key = sel[a], b = a - 1;
        while (b >= 0 && sel[b] > key) { sel[b + 1] = sel[b]; b--; }
        sel[b + 1] = key;
    }
    for (int j = 0; j < 6; j++) idx[bc * 6 + j] = sel[j];
}

__global__ void k_gather(const float* __restrict__ patches, const int* __restrict__ idx,
                         float* __restrict__ yf, bf16_t* __restrict__ yb, int total)
{
    int i = blockIdx.x * 256 + threadIdx.x;
    if (i >= total) return;
    int col = i & 255, row = i >> 8;
    int bc = row / 6, j = row % 6;
    int pi = idx[bc * 6 + j];
    float v = patches[((size_t)bc * 64 + pi) * 256 + col];
    yf[i] = v; yb[i] = to_bf16(v);
}

// out = LN(x [+ r]); dim == blockDim.x (256)
__global__ __launch_bounds__(256) void k_add_ln(
    const float* __restrict__ x, const float* __restrict__ r,
    const float* __restrict__ g, const float* __restrict__ bta,
    float* __restrict__ out, bf16_t* __restrict__ outb, int dim)
{
    const int row = blockIdx.x, t = threadIdx.x;
    __shared__ float red[256];
    float v = x[(size_t)row * dim + t] + (r ? r[(size_t)row * dim + t] : 0.0f);
    red[t] = v; __syncthreads();
    for (int o = 128; o > 0; o >>= 1) { if (t < o) red[t] += red[t + o]; __syncthreads(); }
    float mu = red[0] / (float)dim; __syncthreads();
    red[t] = (v - mu) * (v - mu); __syncthreads();
    for (int o = 128; o > 0; o >>= 1) { if (t < o) red[t] += red[t + o]; __syncthreads(); }
    float inv = rsqrtf(red[0] / (float)dim + 1e-5f);
    float o2 = (v - mu) * inv * g[t] + bta[t];
    out[(size_t)row * dim + t] = o2;
    if (outb) outb[(size_t)row * dim + t] = to_bf16(o2);
}

__global__ void k_scatter(const float* __restrict__ ff, const int* __restrict__ idx,
                          float* __restrict__ combined, int total)
{
    int i = blockIdx.x * 256 + threadIdx.x;
    if (i >= total) return;
    int col = i & 255, row = i >> 8;
    int bc = row / 6, j = row % 6;
    int pi = idx[bc * 6 + j];
    combined[((size_t)bc * 64 + pi) * 256 + col] = ff[i];
}

// overlap-add of 64 patches (stride 8, len 16) into length-520 series
__global__ __launch_bounds__(128) void k_overlap(const float* __restrict__ rec,
                                                 float* __restrict__ acc)
{
    const int bc = blockIdx.x;
    for (int tt = threadIdx.x; tt < 520; tt += 128) {
        int plo = (tt < 16) ? 0 : ((tt - 8) / 8);   // ceil((tt-15)/8)
        int phi = tt / 8; if (phi > 63) phi = 63;
        float s = 0.0f;
        for (int p = plo; p <= phi; p++) {
            int k = tt - p * 8;
            if (k >= 0 && k < 16) s += rec[((size_t)bc * 64 + p) * 16 + k];
        }
        acc[(size_t)bc * 520 + tt] = s;
    }
}

// de-normalize last PRED_LEN=96 steps -> (B=32, 96, C=21)
__global__ void k_final(const float* __restrict__ acc, const float* __restrict__ meanv,
                        const float* __restrict__ stdv, float* __restrict__ out, int total)
{
    int i = blockIdx.x * 256 + threadIdx.x;
    if (i >= total) return;
    int b = i / (96 * 21);
    int rem = i % (96 * 21);
    int tt = rem / 21, c = rem % 21;
    int bc = b * 21 + c;
    int s = 424 + tt;                       // S=520, last 96
    out[i] = acc[(size_t)bc * 520 + s] * stdv[bc] + meanv[bc];
}

// ---------------------------------------------------------------------------
// Host launch
// ---------------------------------------------------------------------------
extern "C" void kernel_launch(void* const* d_in, const int* in_sizes, int n_in,
                              void* d_out, int out_size, void* d_ws, size_t ws_size,
                              hipStream_t stream)
{
    (void)in_sizes; (void)n_in; (void)out_size; (void)ws_size;

    const float* x_enc   = (const float*)d_in[0];
    const float* W_emb   = (const float*)d_in[1];
    const float* scr_W1  = (const float*)d_in[2];
    const float* scr_b1  = (const float*)d_in[3];
    const float* scr_ln_g= (const float*)d_in[4];
    const float* scr_ln_b= (const float*)d_in[5];
    const float* scr_W2  = (const float*)d_in[6];
    const float* scr_b2  = (const float*)d_in[7];
    const float* enc_Wq  = (const float*)d_in[8];
    const float* enc_bq  = (const float*)d_in[9];
    const float* enc_Wk  = (const float*)d_in[10];
    const float* enc_bk  = (const float*)d_in[11];
    const float* enc_Wv  = (const float*)d_in[12];
    const float* enc_bv  = (const float*)d_in[13];
    const float* enc_Wo  = (const float*)d_in[14];
    const float* enc_bo  = (const float*)d_in[15];
    const float* enc_Wf1 = (const float*)d_in[16];
    const float* enc_bf1 = (const float*)d_in[17];
    const float* enc_Wf2 = (const float*)d_in[18];
    const float* enc_bf2 = (const float*)d_in[19];
    const float* enc_n1g = (const float*)d_in[20];
    const float* enc_n1b = (const float*)d_in[21];
    const float* enc_n2g = (const float*)d_in[22];
    const float* enc_n2b = (const float*)d_in[23];
    const float* fin_g   = (const float*)d_in[24];
    const float* fin_b   = (const float*)d_in[25];
    const float* lite_W1 = (const float*)d_in[26];
    const float* lite_b1 = (const float*)d_in[27];
    const float* lite_W2 = (const float*)d_in[28];
    const float* lite_b2 = (const float*)d_in[29];
    const float* reb_W   = (const float*)d_in[30];
    const float* reb_b   = (const float*)d_in[31];

    const int BC = 672, MP = 43008 /*672*64*/, D = 256, M4 = 4032;

    char* ws = (char*)d_ws;
    size_t off = 0;
    auto arena = [&](size_t bytes) -> void* {
        off = (off + 255) & ~(size_t)255;
        void* p = ws + off;
        off += bytes;
        return p;
    };

    float*  meanv     = (float*) arena(BC * 4);
    float*  stdv      = (float*) arena(BC * 4);
    bf16_t* patchA    = (bf16_t*)arena((size_t)MP * 16 * 2);
    float*  patches_f = (float*) arena((size_t)MP * 256 * 4);
    bf16_t* patches_b = (bf16_t*)arena((size_t)MP * 256 * 2);
    float*  h_f       = (float*) arena((size_t)MP * 64 * 4);
    float*  scores    = (float*) arena((size_t)MP * 4);
    int*    idx       = (int*)   arena((size_t)BC * 6 * 4);
    float*  y_f       = (float*) arena((size_t)M4 * 256 * 4);
    bf16_t* y_b       = (bf16_t*)arena((size_t)M4 * 256 * 2);
    bf16_t* q_b       = (bf16_t*)arena((size_t)M4 * 256 * 2);
    bf16_t* k_b       = (bf16_t*)arena((size_t)M4 * 256 * 2);
    bf16_t* v_b       = (bf16_t*)arena((size_t)M4 * 256 * 2);
    float*  tmp_f     = (float*) arena((size_t)M4 * 1024 * 4);
    float*  o_f       = (float*) arena((size_t)M4 * 256 * 4);
    bf16_t* o_b       = (bf16_t*)arena((size_t)M4 * 256 * 2);
    bf16_t* ffh_b     = (bf16_t*)arena((size_t)M4 * 1024 * 2);
    float*  ff_f      = (float*) arena((size_t)M4 * 256 * 4);
    float*  liteh_f   = (float*) arena((size_t)MP * 128 * 4);
    bf16_t* liteh_b   = (bf16_t*)arena((size_t)MP * 128 * 2);
    float*  comb_f    = (float*) arena((size_t)MP * 256 * 4);
    bf16_t* comb_b    = (bf16_t*)arena((size_t)MP * 256 * 2);
    float*  rec_f     = (float*) arena((size_t)MP * 16 * 4);
    float*  accv      = (float*) arena((size_t)BC * 520 * 4);
    bf16_t* wemb_b    = (bf16_t*)arena(16 * 256 * 2);
    bf16_t* scrW1_b   = (bf16_t*)arena(256 * 64 * 2);
    bf16_t* wq_b      = (bf16_t*)arena((size_t)2 * 256 * 256 * 2);
    bf16_t* wk_b      = (bf16_t*)arena((size_t)2 * 256 * 256 * 2);
    bf16_t* wv_b      = (bf16_t*)arena((size_t)2 * 256 * 256 * 2);
    bf16_t* wo_b      = (bf16_t*)arena((size_t)2 * 256 * 256 * 2);
    bf16_t* wf1_b     = (bf16_t*)arena((size_t)2 * 256 * 1024 * 2);
    bf16_t* wf2_b     = (bf16_t*)arena((size_t)2 * 1024 * 256 * 2);
    bf16_t* lw1_b     = (bf16_t*)arena(256 * 128 * 2);
    bf16_t* lw2_b     = (bf16_t*)arena(128 * 256 * 2);
    bf16_t* rw_b      = (bf16_t*)arena(256 * 16 * 2);

    auto conv = [&](const float* s, bf16_t* d, int n) {
        k_f32_to_bf16<<<(n + 255) / 256, 256, 0, stream>>>(s, d, n);
    };
    auto gemm = [&](const bf16_t* A, int lda, const bf16_t* B, int ldb,
                    const float* bias, float* C, int ldc,
                    int M, int N, int K, int act) {
        dim3 g((N + 63) / 64, (M + 63) / 64);
        k_gemm<<<g, 128, 0, stream>>>(A, B, bias, C, M, N, K, lda, ldb, ldc, act);
    };

    // weights -> bf16
    conv(W_emb, wemb_b, 16 * 256);
    conv(scr_W1, scrW1_b, 256 * 64);
    conv(enc_Wq, wq_b, 2 * 256 * 256);
    conv(enc_Wk, wk_b, 2 * 256 * 256);
    conv(enc_Wv, wv_b, 2 * 256 * 256);
    conv(enc_Wo, wo_b, 2 * 256 * 256);
    conv(enc_Wf1, wf1_b, 2 * 256 * 1024);
    conv(enc_Wf2, wf2_b, 2 * 1024 * 256);
    conv(lite_W1, lw1_b, 256 * 128);
    conv(lite_W2, lw2_b, 128 * 256);
    conv(reb_W, rw_b, 256 * 16);

    // RevIN + patch build
    k_prep<<<BC, 256, 0, stream>>>(x_enc, meanv, stdv, patchA);

    // patch embedding + PE
    gemm(patchA, 16, wemb_b, 256, nullptr, patches_f, 256, MP, 256, 16, 0);
    k_add_pe<<<((size_t)MP * 256 + 255) / 256, 256, 0, stream>>>(patches_f, MP * 256);
    conv(patches_f, patches_b, MP * 256);

    // scorer + top-k + gather
    gemm(patches_b, 256, scrW1_b, 64, scr_b1, h_f, 64, MP, 64, 256, 1);
    k_scorer<<<MP, 64, 0, stream>>>(h_f, scr_ln_g, scr_ln_b, scr_W2, scr_b2, scores);
    k_topk<<<(BC + 63) / 64, 64, 0, stream>>>(scores, idx);
    k_gather<<<((size_t)M4 * 256 + 255) / 256, 256, 0, stream>>>(
        patches_f, idx, y_f, y_b, M4 * 256);

    // transformer encoder (2 layers)
    for (int l = 0; l < 2; l++) {
        const size_t wOff = (size_t)l * 256 * 256;
        const size_t fOff = (size_t)l * 256 * 1024;

        gemm(y_b, 256, wq_b + wOff, 256, enc_bq + l * 256, tmp_f, 256, M4, 256, 256, 0);
        conv(tmp_f, q_b, M4 * 256);
        gemm(y_b, 256, wk_b + wOff, 256, enc_bk + l * 256, tmp_f, 256, M4, 256, 256, 0);
        conv(tmp_f, k_b, M4 * 256);
        gemm(y_b, 256, wv_b + wOff, 256, enc_bv + l * 256, tmp_f, 256, M4, 256, 256, 0);
        conv(tmp_f, v_b, M4 * 256);

        dim3 gf(M4 / 64, 8);
        k_flash<<<gf, 128, 0, stream>>>(q_b, k_b, v_b, o_f, M4, D);
        conv(o_f, o_b, M4 * 256);

        gemm(o_b, 256, wo_b + wOff, 256, enc_bo + l * 256, tmp_f, 256, M4, 256, 256, 0);
        k_add_ln<<<M4, 256, 0, stream>>>(tmp_f, y_f, enc_n1g + l * 256,
                                         enc_n1b + l * 256, y_f, y_b, 256);

        gemm(y_b, 256, wf1_b + fOff, 1024, enc_bf1 + l * 1024, tmp_f, 1024,
             M4, 1024, 256, 1);
        conv(tmp_f, ffh_b, M4 * 1024);
        gemm(ffh_b, 1024, wf2_b + fOff, 256, enc_bf2 + l * 256, o_f, 256,
             M4, 256, 1024, 0);
        k_add_ln<<<M4, 256, 0, stream>>>(o_f, y_f, enc_n2g + l * 256,
                                         enc_n2b + l * 256, y_f, y_b, 256);
    }

    // final norm -> full_feat
    k_add_ln<<<M4, 256, 0, stream>>>(y_f, nullptr, fin_g, fin_b, ff_f, nullptr, 256);

    // lite path over all patches
    gemm(patches_b, 256, lw1_b, 128, lite_b1, liteh_f, 128, MP, 128, 256, 1);
    conv(liteh_f, liteh_b, MP * 128);
    gemm(liteh_b, 128, lw2_b, 256, lite_b2, comb_f, 256, MP, 256, 128, 0);

    // combined = lite with top-k rows replaced by full_feat
    k_scatter<<<((size_t)M4 * 256 + 255) / 256, 256, 0, stream>>>(
        ff_f, idx, comb_f, M4 * 256);
    conv(comb_f, comb_b, MP * 256);

    // rebuild + overlap-add + de-norm
    gemm(comb_b, 256, rw_b, 16, reb_b, rec_f, 16, MP, 16, 256, 0);
    k_overlap<<<BC, 128, 0, stream>>>(rec_f, accv);
    k_final<<<(32 * 96 * 21 + 255) / 256, 256, 0, stream>>>(
        accv, meanv, stdv, (float*)d_out, 32 * 96 * 21);
}
